// TorchMD_ET_83004537963111
// MI455X (gfx1250) — matrix operand order repaired and checked by reading, not compiled, not run on hardware
//
#include <hip/hip_runtime.h>
#include <hip/hip_bf16.h>
#include <math.h>

#define N_NODES 20000
#define N_EDGES 640000
#define H       128
#define H3      384
#define NRBF    50
#define EPB     64                         // edges per block (edge kernel)
#define PI_OVER_CUT 0.6283185307179586f    // pi / 5.0

typedef float  v2f   __attribute__((ext_vector_type(2)));
typedef float  v8f   __attribute__((ext_vector_type(8)));
typedef __bf16 v16bf __attribute__((ext_vector_type(16)));

// ---- fp32 WMMA 16x16 tile primitive (V_WMMA_F32_16X16X4_F32) -------------
// A 16x4 f32: lane m=lane&15 holds rows; kh=lane>>4 selects K pair {2kh,2kh+1}
// B 4x16 f32: lane n=lane&15 holds cols; same K-pair selection
// D 16x16 f32: VGPR i -> row (i + 8*kh), col (lane&15)
__device__ __forceinline__ v8f wmma4(v8f acc, v2f a, v2f b) {
    return __builtin_amdgcn_wmma_f32_16x16x4_f32(
        false, a, false, b, (short)0, acc, false, false);
}

// ---- bf16 WMMA (V_WMMA_F32_16X16X32_BF16), fp32 accumulate ---------------
// A 16x32 bf16 fragment element j: j<8 -> K = kh*8+j ; j>=8 -> K = 16+kh*8+j-8
// B 32x16 bf16 fragment element j: j<8 -> K = kh*8+j ; j>=8 -> K = 16+kh*8+j-8, column n = lane&15
__device__ __forceinline__ v8f wmma_bf(v8f acc, v16bf a, v16bf b) {
    return __builtin_amdgcn_wmma_f32_16x16x32_bf16(
        false, a, false, b, (short)0, acc, false, false);
}

__device__ __forceinline__ v2f ld2(const float* p) { return *(const v2f*)p; }

__device__ __forceinline__ float silu_f(float x) {
    return x / (1.0f + __expf(-x));
}

// ===========================================================================
// Kernel 0a: pack a [K,C] f32 weight into K-pair-major [KP][C][2] so an fp32
// WMMA B fragment becomes a single coalesced global_load_b64.
// ===========================================================================
__global__ void pack_w_kernel(const float* __restrict__ W, float* __restrict__ Wp,
                              int K, int C, int KP)
{
    int total = KP * C;
    for (int i = blockIdx.x * blockDim.x + threadIdx.x; i < total;
         i += gridDim.x * blockDim.x) {
        int kp = i / C, c = i % C;
        int k0 = 2 * kp;
        float w0 = (k0     < K) ? W[(size_t)k0 * C + c]       : 0.0f;
        float w1 = (k0 + 1 < K) ? W[(size_t)(k0 + 1) * C + c] : 0.0f;
        Wp[(size_t)i * 2]     = w0;
        Wp[(size_t)i * 2 + 1] = w1;
    }
}

// ===========================================================================
// Kernel 0b: split a [K,C] f32 weight (K padded to 64) into hi/lo bf16 in
// B-fragment order: flat = ((kc2*C + c)*16 + j), K = (kc2>>1)*32 + (j<8 ? (kc2&1)*8+j : 16+(kc2&1)*8+j-8).
// One lane's B fragment = 32 contiguous bytes (coalesced across lanes).
// ===========================================================================
__global__ void pack_bf16_kernel(const float* __restrict__ W,
                                 __bf16* __restrict__ Whi, __bf16* __restrict__ Wlo,
                                 int K, int C)
{
    int total = 4 * C * 16;
    for (int i = blockIdx.x * blockDim.x + threadIdx.x; i < total;
         i += gridDim.x * blockDim.x) {
        int kc2 = i / (C * 16);
        int c   = (i >> 4) % C;
        int j   = i & 15;
        int Kk  = (kc2 >> 1) * 32 + ((j < 8) ? ((kc2 & 1) * 8 + j) : (16 + (kc2 & 1) * 8 + j - 8));
        float w = (Kk < K) ? W[(size_t)Kk * C + c] : 0.0f;
        __bf16 h = (__bf16)w;
        Whi[i] = h;
        Wlo[i] = (__bf16)(w - (float)h);
    }
}

// ===========================================================================
// Kernel 1: per 16-node tile: LayerNorm + q/k/v GEMMs + vp = vec @ vecw.
// (fp32 WMMA; q/k share A fragments, vp shares B across 3 spatial row tiles)
// ===========================================================================
__global__ void __launch_bounds__(256)
node_pre_kernel(const float* __restrict__ x,    const float* __restrict__ vec,
                const float* __restrict__ lng,  const float* __restrict__ lnb,
                const float* __restrict__ qwP,  const float* __restrict__ qb,
                const float* __restrict__ kwP,  const float* __restrict__ kb,
                const float* __restrict__ vwP,  const float* __restrict__ vb,
                const float* __restrict__ vecwP,
                float* __restrict__ qo, float* __restrict__ ko,
                float* __restrict__ vo, float* __restrict__ vdot,
                float* __restrict__ vec3o)
{
    __shared__ float xn[16][H];
    __shared__ float vp1[48][H];
    __shared__ float vp2[48][H];
    __shared__ float mu_s[16], rs_s[16];

    const int tid   = threadIdx.x;
    const int node0 = blockIdx.x * 16;

    for (int i = tid; i < 16 * H; i += 256)
        xn[i >> 7][i & 127] = x[(size_t)node0 * H + i];
    __syncthreads();
    if (tid < 16) {
        float s = 0.f, s2 = 0.f;
        for (int c = 0; c < H; ++c) { float t = xn[tid][c]; s += t; s2 += t * t; }
        float mu  = s * (1.0f / H);
        float var = s2 * (1.0f / H) - mu * mu;
        mu_s[tid] = mu;
        rs_s[tid] = rsqrtf(var + 1e-5f);
    }
    __syncthreads();
    for (int i = tid; i < 16 * H; i += 256) {
        int r = i >> 7, c = i & 127;
        xn[r][c] = (xn[r][c] - mu_s[r]) * rs_s[r] * lng[c] + lnb[c];
    }
    __syncthreads();

    const int lane = tid & 31, wid = tid >> 5;
    const int m = lane & 15, kh = lane >> 4, n = lane & 15;
    const v2f* qp = (const v2f*)qwP;
    const v2f* kp = (const v2f*)kwP;
    const v2f* vp = (const v2f*)vwP;
    const v2f* wp = (const v2f*)vecwP;

    { // q & k fused (shared A fragment)
        const int cb = wid * 16;
        v8f accq = {}, acck = {};
        #pragma unroll
        for (int k0 = 0; k0 < H; k0 += 4) {
            int kp2 = (k0 >> 1) + kh;
            v2f a  = ld2(&xn[m][k0 + 2 * kh]);
            v2f bq = qp[kp2 * H + cb + n];
            v2f bk = kp[kp2 * H + cb + n];
            accq = wmma4(accq, a, bq);
            acck = wmma4(acck, a, bk);
        }
        #pragma unroll
        for (int i = 0; i < 8; ++i) {
            int row = i + 8 * kh, col = cb + n;
            qo[(size_t)(node0 + row) * H + col] = accq[i] + qb[col];
            ko[(size_t)(node0 + row) * H + col] = acck[i] + kb[col];
        }
    }
    for (int t = wid; t < 24; t += 8) { // v tiles
        const int cb = t * 16;
        v8f acc = {};
        #pragma unroll
        for (int k0 = 0; k0 < H; k0 += 4) {
            int kp2 = (k0 >> 1) + kh;
            v2f a = ld2(&xn[m][k0 + 2 * kh]);
            v2f b = vp[kp2 * H3 + cb + n];
            acc = wmma4(acc, a, b);
        }
        #pragma unroll
        for (int i = 0; i < 8; ++i) {
            int row = i + 8 * kh, col = cb + n;
            vo[(size_t)(node0 + row) * H3 + col] = acc[i] + vb[col];
        }
    }
    // vp = vec @ vecw : one col tile per wave, 3 spatial row tiles share B
    const float* va = vec + (size_t)node0 * H3;
    for (int ct = wid; ct < 24; ct += 8) {
        const int cb = ct * 16;
        v8f acc0 = {}, acc1 = {}, acc2 = {};
        #pragma unroll
        for (int k0 = 0; k0 < H; k0 += 4) {
            int kp2 = (k0 >> 1) + kh;
            int kk  = k0 + 2 * kh;
            v2f b  = wp[kp2 * H3 + cb + n];
            v2f a0 = ld2(&va[(     m) * H + kk]);
            v2f a1 = ld2(&va[(16 + m) * H + kk]);
            v2f a2 = ld2(&va[(32 + m) * H + kk]);
            acc0 = wmma4(acc0, a0, b);
            acc1 = wmma4(acc1, a1, b);
            acc2 = wmma4(acc2, a2, b);
        }
        #pragma unroll
        for (int i = 0; i < 8; ++i) {
            int row = i + 8 * kh;
            if (ct < 8) {
                vp1[     row][cb + n] = acc0[i];
                vp1[16 + row][cb + n] = acc1[i];
                vp1[32 + row][cb + n] = acc2[i];
            } else if (ct < 16) {
                vp2[     row][cb - 128 + n] = acc0[i];
                vp2[16 + row][cb - 128 + n] = acc1[i];
                vp2[32 + row][cb - 128 + n] = acc2[i];
            } else {
                int col = cb - 256 + n;
                vec3o[((size_t)node0 * 3 +      row) * H + col] = acc0[i];
                vec3o[((size_t)node0 * 3 + 16 + row) * H + col] = acc1[i];
                vec3o[((size_t)node0 * 3 + 32 + row) * H + col] = acc2[i];
            }
        }
    }
    __syncthreads();

    for (int i = tid; i < 16 * H; i += 256) {
        int nl = i >> 7, c = i & 127;
        int r = nl * 3;
        float s = vp1[r][c] * vp2[r][c]
                + vp1[r + 1][c] * vp2[r + 1][c]
                + vp1[r + 2][c] * vp2[r + 2][c];
        vdot[(size_t)(node0 + nl) * H + c] = s;
    }
}

// ===========================================================================
// Kernel 2: zero the aggregation buffers (x_agg ++ vec_agg, contiguous)
// ===========================================================================
__global__ void zero_kernel(float* __restrict__ p, size_t count)
{
    size_t i = (size_t)blockIdx.x * blockDim.x + threadIdx.x;
    size_t stride = (size_t)gridDim.x * blockDim.x;
    for (; i < count; i += stride) p[i] = 0.0f;
}

// ===========================================================================
// Kernel 3: per 64-edge tile (dynamic LDS ~147 KB):
//   - f_ij split into hi/lo bf16 in A-fragment order in LDS (K padded to 64)
//   - dk/dv GEMM via bf16x3 emulation on v_wmma_f32_16x16x32_bf16:
//     D += Ah*Bh + Ah*Bl + Al*Bh (fp32 accumulate, ~fp32 accuracy)
//     each wave: 2 column tiles x 4 edge row-tiles (8 f32 accumulators)
//   - SiLU epilogue into LDS, per-head attention + cosine cutoff,
//     atomic scatter-add of xm / vecm into node aggregates (L2-resident).
// v/dv are head-interleaved: head h owns flat [h*48, h*48+48) split 16/16/16.
// ===========================================================================
__global__ void __launch_bounds__(256)
edge_kernel(const float* __restrict__ f_ij, const int* __restrict__ ei,
            const float* __restrict__ r_ij, const float* __restrict__ d_ij,
            const float* __restrict__ vec,
            const __bf16* __restrict__ dkHi, const __bf16* __restrict__ dkLo,
            const __bf16* __restrict__ dvHi, const __bf16* __restrict__ dvLo,
            const float* __restrict__ dkb,  const float* __restrict__ dvb,
            const float* __restrict__ qn,  const float* __restrict__ kn,
            const float* __restrict__ vn,
            float* __restrict__ xagg, float* __restrict__ vagg)
{
    extern __shared__ float smem[];
    float*  dkv   = smem;                         // [EPB][512] f32 dk|dv
    __bf16* fa_hi = (__bf16*)(dkv + EPB * 512);   // [4][EPB] A fragments (16 bf16)
    __bf16* fa_lo = fa_hi + 4 * EPB * 16;
    float*  attn_s = (float*)(fa_lo + 4 * EPB * 16);  // [EPB][8]
    float*  r_s    = attn_s + EPB * 8;
    int*    src_s  = (int*)(r_s + EPB);
    int*    dst_s  = src_s + EPB;

    const int tid = threadIdx.x;
    const int e0  = blockIdx.x * EPB;

    if (tid < EPB) {
        src_s[tid] = ei[e0 + tid];
        dst_s[tid] = ei[N_EDGES + e0 + tid];
        r_s[tid]   = r_ij[e0 + tid];
    }
    // stage f_ij hi/lo into A-fragment order: flat = ((kc2*EPB + row)*16 + j)
    for (int i = tid; i < 4 * EPB * 16; i += 256) {
        int kc2 = i >> 10;                 // /(EPB*16)
        int row = (i >> 4) & (EPB - 1);
        int j   = i & 15;
        int kc  = kc2 >> 1, khf = kc2 & 1;
        int Kk  = kc * 32 + ((j < 8) ? (khf * 8 + j) : (16 + khf * 8 + j - 8));
        float w = (Kk < NRBF) ? f_ij[(size_t)(e0 + row) * NRBF + Kk] : 0.0f;
        __bf16 h = (__bf16)w;
        fa_hi[i] = h;
        fa_lo[i] = (__bf16)(w - (float)h);
    }
    __syncthreads();

    const int lane = tid & 31, wid = tid >> 5;
    const int m = lane & 15, kh = lane >> 4, n = lane & 15;
    const v16bf* FH  = (const v16bf*)fa_hi;   // fragment index kc2*EPB + row
    const v16bf* FL  = (const v16bf*)fa_lo;
    const v16bf* DKH = (const v16bf*)dkHi;    // fragment index kc2*H  + c
    const v16bf* DKL = (const v16bf*)dkLo;
    const v16bf* DVH = (const v16bf*)dvHi;    // fragment index kc2*H3 + c
    const v16bf* DVL = (const v16bf*)dvLo;

    // 32 col-tiles (8 dk + 24 dv); wave handles pair (t, t+16), 2 iterations
    for (int t = wid; t < 16; t += 8) {
        const bool    xdk = (t < 8);
        const int     cbX = xdk ? t * 16 : (t - 8) * 16;
        const v16bf*  BXH = xdk ? DKH : DVH;
        const v16bf*  BXL = xdk ? DKL : DVL;
        const int     CX  = xdk ? H : H3;
        const float*  BbX = xdk ? dkb : dvb;
        const int     cbY = (t + 8) * 16;       // tile t+16 is always dv
        v8f x0 = {}, x1 = {}, x2 = {}, x3 = {};
        v8f y0 = {}, y1 = {}, y2 = {}, y3 = {};
        #pragma unroll
        for (int kc = 0; kc < 2; ++kc) {
            const int kc2 = kc * 2 + kh;
            v16bf bxh = BXH[kc2 * CX + cbX + n];
            v16bf bxl = BXL[kc2 * CX + cbX + n];
            v16bf byh = DVH[kc2 * H3 + cbY + n];
            v16bf byl = DVL[kc2 * H3 + cbY + n];
            {
                v16bf ah = FH[kc2 * EPB + m], al = FL[kc2 * EPB + m];
                x0 = wmma_bf(x0, ah, bxh); x0 = wmma_bf(x0, ah, bxl); x0 = wmma_bf(x0, al, bxh);
                y0 = wmma_bf(y0, ah, byh); y0 = wmma_bf(y0, ah, byl); y0 = wmma_bf(y0, al, byh);
            }
            {
                v16bf ah = FH[kc2 * EPB + 16 + m], al = FL[kc2 * EPB + 16 + m];
                x1 = wmma_bf(x1, ah, bxh); x1 = wmma_bf(x1, ah, bxl); x1 = wmma_bf(x1, al, bxh);
                y1 = wmma_bf(y1, ah, byh); y1 = wmma_bf(y1, ah, byl); y1 = wmma_bf(y1, al, byh);
            }
            {
                v16bf ah = FH[kc2 * EPB + 32 + m], al = FL[kc2 * EPB + 32 + m];
                x2 = wmma_bf(x2, ah, bxh); x2 = wmma_bf(x2, ah, bxl); x2 = wmma_bf(x2, al, bxh);
                y2 = wmma_bf(y2, ah, byh); y2 = wmma_bf(y2, ah, byl); y2 = wmma_bf(y2, al, byh);
            }
            {
                v16bf ah = FH[kc2 * EPB + 48 + m], al = FL[kc2 * EPB + 48 + m];
                x3 = wmma_bf(x3, ah, bxh); x3 = wmma_bf(x3, ah, bxl); x3 = wmma_bf(x3, al, bxh);
                y3 = wmma_bf(y3, ah, byh); y3 = wmma_bf(y3, ah, byl); y3 = wmma_bf(y3, al, byh);
            }
        }
        const float biasX = BbX[cbX + n];
        const float biasY = dvb[cbY + n];
        const int   colX  = (xdk ? 0 : H) + cbX + n;
        const int   colY  = H + cbY + n;
        #pragma unroll
        for (int i = 0; i < 8; ++i) {
            int row = i + 8 * kh;
            dkv[(     row) * 512 + colX] = silu_f(x0[i] + biasX);
            dkv[(16 + row) * 512 + colX] = silu_f(x1[i] + biasX);
            dkv[(32 + row) * 512 + colX] = silu_f(x2[i] + biasX);
            dkv[(48 + row) * 512 + colX] = silu_f(x3[i] + biasX);
            dkv[(     row) * 512 + colY] = silu_f(y0[i] + biasY);
            dkv[(16 + row) * 512 + colY] = silu_f(y1[i] + biasY);
            dkv[(32 + row) * 512 + colY] = silu_f(y2[i] + biasY);
            dkv[(48 + row) * 512 + colY] = silu_f(y3[i] + biasY);
        }
    }
    __syncthreads();

    // ---- attention: one (edge, head) per thread, 2 rounds ----
    for (int p = tid; p < EPB * 8; p += 256) {
        int e = p >> 3, h = p & 7;
        int s = src_s[e], d = dst_s[e];
        float acc = 0.f;
        #pragma unroll
        for (int dd = 0; dd < 16; ++dd) {
            int c = h * 16 + dd;
            acc += qn[(size_t)d * H + c] * kn[(size_t)s * H + c] * dkv[e * 512 + c];
        }
        acc = silu_f(acc);
        float r   = r_s[e];
        float cut = (r < 5.0f) ? 0.5f * (__cosf(r * PI_OVER_CUT) + 1.0f) : 0.0f;
        attn_s[e * 8 + h] = acc * cut;
    }
    __syncthreads();

    // ---- scatter-add: EPB edges x 512 outputs (128 xm + 384 vecm) ----
    for (int w = tid; w < EPB * 512; w += 256) {
        int e = w >> 9, j = w & 511;
        int s = src_s[e], d = dst_s[e];
        if (j < 128) {
            int h = j >> 4, dd = j & 15;
            int fc = h * 48 + dd;                          // xm chunk of head h
            float val = vn[(size_t)s * H3 + fc] * dkv[e * 512 + H + fc]
                      * attn_s[e * 8 + h];
            atomicAdd(&xagg[(size_t)d * H + j], val);
        } else {
            int jj = j - 128;
            int ax = jj >> 7, cc = jj & 127;
            int h = cc >> 4, dd = cc & 15;
            int f1 = h * 48 + 16 + dd;                     // vec1m chunk
            int f2 = h * 48 + 32 + dd;                     // vec2m chunk
            float v1m = vn[(size_t)s * H3 + f1] * dkv[e * 512 + H + f1];
            float v2m = vn[(size_t)s * H3 + f2] * dkv[e * 512 + H + f2];
            float val = vec[(size_t)s * H3 + ax * H + cc] * v1m
                      + v2m * d_ij[(size_t)(e0 + e) * 3 + ax];
            atomicAdd(&vagg[(size_t)d * H3 + jj], val);
        }
    }
}

// ===========================================================================
// Kernel 4 (32 nodes/block): o = x_agg @ ow + ob with 2 row-tile accumulators
// per B-fragment load; then dx = vec_dot*o2 + o3 ; dvec = vec3*o1 + vec_agg.
// ===========================================================================
__global__ void __launch_bounds__(256)
node_out_kernel(const float* __restrict__ xagg, const float* __restrict__ vagg,
                const float* __restrict__ vdot, const float* __restrict__ vec3,
                const float* __restrict__ owP,  const float* __restrict__ ob,
                float* __restrict__ dx, float* __restrict__ dvec)
{
    __shared__ float o_s[32][H3];      // 48 KB
    const int tid   = threadIdx.x;
    const int node0 = blockIdx.x * 32;
    const int lane = tid & 31, wid = tid >> 5;
    const int m = lane & 15, kh = lane >> 4, n = lane & 15;
    const float* A = xagg + (size_t)node0 * H;
    const v2f* op = (const v2f*)owP;

    for (int t = wid; t < 24; t += 8) {
        const int cb = t * 16;
        v8f acc0 = {}, acc1 = {};
        #pragma unroll
        for (int k0 = 0; k0 < H; k0 += 4) {
            int kp2 = (k0 >> 1) + kh;
            int kk  = k0 + 2 * kh;
            v2f b  = op[kp2 * H3 + cb + n];
            v2f a0 = ld2(&A[(     m) * H + kk]);
            v2f a1 = ld2(&A[(16 + m) * H + kk]);
            acc0 = wmma4(acc0, a0, b);
            acc1 = wmma4(acc1, a1, b);
        }
        #pragma unroll
        for (int i = 0; i < 8; ++i) {
            o_s[     i + 8 * kh][cb + n] = acc0[i] + ob[cb + n];
            o_s[16 + i + 8 * kh][cb + n] = acc1[i] + ob[cb + n];
        }
    }
    __syncthreads();

    for (int i = tid; i < 32 * H; i += 256) {
        int nl = i >> 7, c = i & 127;
        dx[(size_t)(node0 + nl) * H + c] =
            vdot[(size_t)(node0 + nl) * H + c] * o_s[nl][H + c] + o_s[nl][2 * H + c];
    }
    for (int i = tid; i < 32 * H3; i += 256) {
        int nl = i / H3, j = i % H3;
        int c = j & 127;
        dvec[(size_t)(node0 + nl) * H3 + j] =
            vec3[((size_t)(node0 + nl) * 3 + (j >> 7)) * H + c] * o_s[nl][c]
            + vagg[(size_t)(node0 + nl) * H3 + j];
    }
}

// ===========================================================================
extern "C" void kernel_launch(void* const* d_in, const int* in_sizes, int n_in,
                              void* d_out, int out_size, void* d_ws, size_t ws_size,
                              hipStream_t stream)
{
    const float* x    = (const float*)d_in[0];
    const float* vec  = (const float*)d_in[1];
    const int*   ei   = (const int*)  d_in[2];
    const float* r_ij = (const float*)d_in[3];
    const float* f_ij = (const float*)d_in[4];
    const float* d_ij = (const float*)d_in[5];
    const float* lng  = (const float*)d_in[6];
    const float* lnb  = (const float*)d_in[7];
    const float* qw   = (const float*)d_in[8];
    const float* qb   = (const float*)d_in[9];
    const float* kw   = (const float*)d_in[10];
    const float* kb   = (const float*)d_in[11];
    const float* vw   = (const float*)d_in[12];
    const float* vb   = (const float*)d_in[13];
    const float* ow   = (const float*)d_in[14];
    const float* ob   = (const float*)d_in[15];
    const float* vecw = (const float*)d_in[16];
    const float* dkw  = (const float*)d_in[17];
    const float* dkb  = (const float*)d_in[18];
    const float* dvw  = (const float*)d_in[19];
    const float* dvb  = (const float*)d_in[20];

    // ---- workspace layout (floats) ----
    float* ws    = (float*)d_ws;
    float* q_ws  = ws;                               // [N, H]
    float* k_ws  = ws + (size_t)N_NODES * 1 * H;     // [N, H]
    float* v_ws  = ws + (size_t)N_NODES * 2 * H;     // [N, 3H]
    float* vdot  = ws + (size_t)N_NODES * 5 * H;     // [N, H]
    float* vec3w = ws + (size_t)N_NODES * 6 * H;     // [N, 3, H]
    float* xagg  = ws + (size_t)N_NODES * 9 * H;     // [N, H]   (contiguous with
    float* vagg  = ws + (size_t)N_NODES * 10 * H;    // [N, 3H]   vagg for zeroing)

    float* wbase = ws + (size_t)N_NODES * 13 * H;
    float* qwP   = wbase;                            // 64*128*2  = 16384 f32
    float* kwP   = qwP  + 64 * H  * 2;
    float* vwP   = kwP  + 64 * H  * 2;               // 64*384*2  = 49152 f32
    float* vecwP = vwP  + 64 * H3 * 2;
    float* owP   = vecwP + 64 * H3 * 2;
    __bf16* dkHi = (__bf16*)(owP + 64 * H3 * 2);     // 4*128*16 bf16
    __bf16* dkLo = dkHi + 4 * H  * 16;
    __bf16* dvHi = dkLo + 4 * H  * 16;               // 4*384*16 bf16
    __bf16* dvLo = dvHi + 4 * H3 * 16;

    float* dx   = (float*)d_out;                     // [N, H]
    float* dvec = dx + (size_t)N_NODES * H;          // [N, 3, H]

    const size_t edge_lds = (size_t)(EPB * 512 + EPB * 8 + EPB) * sizeof(float)
                          + (size_t)(2 * 4 * EPB * 16) * sizeof(__bf16)
                          + (size_t)(2 * EPB) * sizeof(int);

    // ---- pack weights (fp32 pair-major for node GEMMs; bf16 hi/lo for edge) ----
    pack_w_kernel<<<32, 256, 0, stream>>>(qw,   qwP,   H, H,  64);
    pack_w_kernel<<<32, 256, 0, stream>>>(kw,   kwP,   H, H,  64);
    pack_w_kernel<<<64, 256, 0, stream>>>(vw,   vwP,   H, H3, 64);
    pack_w_kernel<<<64, 256, 0, stream>>>(vecw, vecwP, H, H3, 64);
    pack_w_kernel<<<64, 256, 0, stream>>>(ow,   owP,   H, H3, 64);
    pack_bf16_kernel<<<16, 256, 0, stream>>>(dkw, dkHi, dkLo, NRBF, H);
    pack_bf16_kernel<<<32, 256, 0, stream>>>(dvw, dvHi, dvLo, NRBF, H3);

    zero_kernel<<<2048, 256, 0, stream>>>(xagg, (size_t)N_NODES * 4 * H);
    node_pre_kernel<<<N_NODES / 16, 256, 0, stream>>>(
        x, vec, lng, lnb, qwP, qb, kwP, kb, vwP, vb, vecwP,
        q_ws, k_ws, v_ws, vdot, vec3w);
    edge_kernel<<<N_EDGES / EPB, 256, edge_lds, stream>>>(
        f_ij, ei, r_ij, d_ij, vec, dkHi, dkLo, dvHi, dvLo, dkb, dvb,
        q_ws, k_ws, v_ws, xagg, vagg);
    node_out_kernel<<<N_NODES / 32, 256, 0, stream>>>(
        xagg, vagg, vdot, vec3w, owP, ob, dx, dvec);
}
